// VectorQuantizer_41162966565648
// MI455X (gfx1250) — matrix operand order repaired and checked by reading, not compiled, not run on hardware
//
#include <hip/hip_runtime.h>
#include <stdint.h>

#define DIM    256
#define HW     1024            // 32*32
#define NROWS  32768           // 32 batches * 1024 hw
#define NCODES 1024
#define NDTOT  (NROWS * DIM)   // 8388608

typedef __attribute__((ext_vector_type(16))) __bf16 v16bf;
typedef __attribute__((ext_vector_type(8)))  float  v8f;

#define WMMA_BF16(A, B, C) \
  __builtin_amdgcn_wmma_f32_16x16x32_bf16(false, (A), false, (B), (short)0, (C), false, false)

// ---------- helpers ----------
__device__ __forceinline__ unsigned short f2bf(float f) {   // RNE fp32->bf16
  unsigned u = __float_as_uint(f);
  u += 0x7fffu + ((u >> 16) & 1u);
  return (unsigned short)(u >> 16);
}
__device__ __forceinline__ float bf2f(unsigned short h) {
  return __uint_as_float(((unsigned)h) << 16);
}
__device__ __forceinline__ v16bf frag(uint4 a, uint4 b) {
  union { uint4 q[2]; v16bf v; } u;
  u.q[0] = a; u.q[1] = b;
  return u.v;
}
__device__ __forceinline__ unsigned ordf(float v) {         // monotone f32->u32
  unsigned u = __float_as_uint(v);
  return (u & 0x80000000u) ? ~u : (u | 0x80000000u);
}
__device__ __forceinline__ unsigned long long shflx64(unsigned long long p, int m) {
  unsigned lo = (unsigned)__shfl_xor((int)(unsigned)p, m, 32);
  unsigned hi = (unsigned)__shfl_xor((int)(p >> 32), m, 32);
  return (((unsigned long long)hi) << 32) | lo;
}
// CDNA5 async global->LDS copy (ASYNCcnt-tracked DMA into LDS)
__device__ __forceinline__ void async_b128(unsigned lds_off, const void* gaddr) {
  asm volatile("global_load_async_to_lds_b128 %0, %1, off"
               :: "v"(lds_off), "v"((unsigned long long)(size_t)gaddr)
               : "memory");
}
__device__ __forceinline__ void wait_async0() {
  asm volatile("s_wait_asynccnt 0x0" ::: "memory");
}

// ---------- phase 1a: weight -> bf16 hi/lo + 0.5*||w||^2, zero winners ----------
__global__ void vq_prep_w(const float* __restrict__ w,
                          unsigned short* __restrict__ whi,
                          unsigned short* __restrict__ wlo,
                          float* __restrict__ hwn,
                          unsigned long long* __restrict__ best) {
  int k = blockIdx.x, t = threadIdx.x;             // 1024 blocks x 256 threads
  float v = w[k * DIM + t];
  unsigned short h = f2bf(v);
  whi[k * DIM + t] = h;
  wlo[k * DIM + t] = f2bf(v - bf2f(h));
  __shared__ float red[256];
  red[t] = v * v;
  __syncthreads();
  for (int s = 128; s > 0; s >>= 1) { if (t < s) red[t] += red[t + s]; __syncthreads(); }
  if (t == 0) hwn[k] = 0.5f * red[0];
  if (t < 32) best[k * 32 + t] = 0ULL;             // 1024*32 == NROWS
}

// ---------- phase 1b: NCHW fp32 -> row-major [N,256] bf16 hi/lo ----------
__global__ void vq_prep_x(const float* __restrict__ in,
                          unsigned short* __restrict__ xhi,
                          unsigned short* __restrict__ xlo) {
  __shared__ __align__(16) unsigned short lh[32 * 264];   // padded rows (528B = 33*16)
  __shared__ __align__(16) unsigned short ll[32 * 264];
  int g = blockIdx.x;              // 1024 blocks: (b, hw-chunk of 32)
  int b = g >> 5;
  int hw0 = (g & 31) << 5;
  int t = threadIdx.x;
  int hwl = t & 31;                // lane -> consecutive hw: coalesced reads
  int dg  = t >> 5;                // 8 d-groups
  const float* src = in + (size_t)b * DIM * HW + hw0 + hwl;
  for (int j = 0; j < 32; ++j) {
    int d = dg * 32 + j;
    float v = src[(size_t)d * HW];
    unsigned short h = f2bf(v);
    lh[hwl * 264 + d] = h;
    ll[hwl * 264 + d] = f2bf(v - bf2f(h));
  }
  __syncthreads();
  uint4* gh = (uint4*)xhi;
  uint4* gl = (uint4*)xlo;
  for (int u = t; u < 1024; u += 256) {            // 32 rows x 32 uint4
    int row = u >> 5, part = u & 31;
    const uint4* sh = (const uint4*)(lh + row * 264);
    const uint4* sl = (const uint4*)(ll + row * 264);
    size_t gi = (size_t)(b * HW + hw0 + row) * 32 + part;
    gh[gi] = sh[part];
    gl[gi] = sl[part];
  }
}

// ---------- phase 2: split-bf16 WMMA GEMM + fused argmax ----------
__global__ void __launch_bounds__(256)
vq_gemm(const unsigned short* __restrict__ xhi,
        const unsigned short* __restrict__ xlo,
        const unsigned short* __restrict__ whi,
        const unsigned short* __restrict__ wlo,
        const float* __restrict__ hwn,
        unsigned long long* __restrict__ best) {
  // double-buffered W tile: 128 codes x 32 d bf16 (hi+lo), rows padded 4->5 uint4
  __shared__ uint4 Bh[2][128 * 5];
  __shared__ uint4 Bl[2][128 * 5];
  int t = threadIdx.x;
  int lane = t & 31, wv = t >> 5;
  int nbase = blockIdx.x * 128;    // code tile base
  int mbase = blockIdx.y * 128;    // row tile base
  int m0 = mbase + wv * 16;        // this wave's 16 rows

  v8f acc[8] = {};                 // 8 code tiles of 16, f32 accum

  const uint4* gwh = (const uint4*)whi;
  const uint4* gwl = (const uint4*)wlo;
  const uint4* gxh = (const uint4*)xhi;
  const uint4* gxl = (const uint4*)xlo;

  unsigned bh_base = (unsigned)(size_t)(void*)&Bh[0][0];   // low 32 bits = LDS offset
  unsigned bl_base = (unsigned)(size_t)(void*)&Bl[0][0];

  int arow = m0 + (lane & 15);
  int ab = (lane < 16) ? 0 : 8;    // ISA 16-bit A layout: K chunks {0..7,16..23}/{8..15,24..31}

  // async stage of W tile for step s2 into buffer buf
  auto stage = [&](int s2, int buf) {
    int k0q = s2 * 4;                              // k0/8 in uint4 units
    for (int u = t; u < 512; u += 256) {
      int row = u >> 2, part = u & 3;
      size_t gi = (size_t)(nbase + row) * 32 + k0q + part;
      unsigned loff = (unsigned)((buf * 640 + row * 5 + part) * 16);
      async_b128(bh_base + loff, gwh + gi);
      async_b128(bl_base + loff, gwl + gi);
    }
  };

  stage(0, 0);                                     // prologue: DMA first tile
  size_t a0 = (size_t)arow * 32 + (ab >> 3);
  v16bf ah = frag(gxh[a0], gxh[a0 + 2]);           // d1..d1+7, d1+16..d1+23
  v16bf al = frag(gxl[a0], gxl[a0 + 2]);

  for (int s = 0; s < 8; ++s) {                    // D = 256 in steps of 32
    wait_async0();                                 // own async copies done
    __syncthreads();                               // everyone's copies done
    if (s + 1 < 8) stage(s + 1, (s + 1) & 1);      // DMA next tile during compute
    v16bf ahn = ah, aln = al;
    if (s + 1 < 8) {                               // prefetch next A fragments
      size_t a1 = (size_t)arow * 32 + (((s + 1) * 32 + ab) >> 3);
      ahn = frag(gxh[a1], gxh[a1 + 2]);
      aln = frag(gxl[a1], gxl[a1 + 2]);
    }
    const uint4* Bhc = Bh[s & 1];
    const uint4* Blc = Bl[s & 1];
    for (int tp = 0; tp < 4; ++tp) {               // tiles in pairs: independent accums
      int t0 = tp * 2, t1 = t0 + 1;
      int off0 = (t0 * 16 + (lane & 15)) * 5 + (lane >> 4);
      int off1 = (t1 * 16 + (lane & 15)) * 5 + (lane >> 4);
      v16bf bh0 = frag(Bhc[off0], Bhc[off0 + 2]);
      v16bf bh1 = frag(Bhc[off1], Bhc[off1 + 2]);
      v16bf bl0 = frag(Blc[off0], Blc[off0 + 2]);
      v16bf bl1 = frag(Blc[off1], Blc[off1 + 2]);
      // dot ~= xh*wh + xh*wl + xl*wh, interleaved across the two tiles
      acc[t0] = WMMA_BF16(ah, bh0, acc[t0]);
      acc[t1] = WMMA_BF16(ah, bh1, acc[t1]);
      acc[t0] = WMMA_BF16(ah, bl0, acc[t0]);
      acc[t1] = WMMA_BF16(ah, bl1, acc[t1]);
      acc[t0] = WMMA_BF16(al, bh0, acc[t0]);
      acc[t1] = WMMA_BF16(al, bh1, acc[t1]);
    }
    ah = ahn; al = aln;
  }

  float hw8[8];
  for (int ti = 0; ti < 8; ++ti) hw8[ti] = hwn[nbase + ti * 16 + (lane & 15)];

  // argmax of (dot - 0.5||w||^2); pack {ordered(score), ~code} so max == first-min of dist
  for (int r = 0; r < 8; ++r) {    // C layout: VGPR r -> row r + 8*(lane/16)
    unsigned long long p = 0ULL;
    for (int ti = 0; ti < 8; ++ti) {
      float v = acc[ti][r] - hw8[ti];
      unsigned n = (unsigned)(nbase + ti * 16 + (lane & 15));
      unsigned long long q = (((unsigned long long)ordf(v)) << 32) | (unsigned)(~n);
      p = (q > p) ? q : p;
    }
    for (int m = 1; m < 16; m <<= 1) {             // reduce across the 16-lane half
      unsigned long long q = shflx64(p, m);
      p = (q > p) ? q : p;
    }
    if ((lane & 15) == 0) {
      int mrow = mbase + wv * 16 + r + ((lane >> 4) << 3);
      atomicMax(best + mrow, p);                   // total order -> deterministic
    }
  }
}

// ---------- phase 3: gather codes, write NCHW output, loss partials ----------
__global__ void vq_out(const float* __restrict__ in,
                       const float* __restrict__ w,
                       const unsigned long long* __restrict__ best,
                       float* __restrict__ out,
                       float* __restrict__ part) {
  int g = blockIdx.x;              // b*32 + h
  int b = g >> 5, h = g & 31;
  int t = threadIdx.x;
  int wcol = t & 31, dg = t >> 5;
  int n = b * HW + h * 32 + wcol;
  unsigned code = ~(unsigned)(best[n] & 0xffffffffULL);
  const float* wr = w + (size_t)code * DIM;
  float lsum = 0.f;
  for (int j = 0; j < 32; ++j) {
    int d = dg * 32 + j;
    size_t gi = ((size_t)(b * DIM + d)) * HW + h * 32 + wcol;  // coalesced over wcol
    float q = wr[d];
    float x = in[gi];
    float df = q - x;
    lsum += df * df;
    out[gi] = q;                   // straight-through value == quantized
  }
  __shared__ float red[256];
  red[t] = lsum; __syncthreads();
  for (int s = 128; s > 0; s >>= 1) { if (t < s) red[t] += red[t + s]; __syncthreads(); }
  if (t == 0) part[g] = red[0];
}

// ---------- phase 4: deterministic fixed-order loss reduction ----------
__global__ void vq_loss(const float* __restrict__ part, float* __restrict__ loss_slot) {
  __shared__ float red[256];
  int t = threadIdx.x;
  float s = 0.f;
  for (int i = t; i < 1024; i += 256) s += part[i];
  red[t] = s; __syncthreads();
  for (int k = 128; k > 0; k >>= 1) { if (t < k) red[t] += red[t + k]; __syncthreads(); }
  if (t == 0) loss_slot[0] = red[0] * (1.25f / (float)NDTOT);  // codebook + 0.25*commit
}

extern "C" void kernel_launch(void* const* d_in, const int* in_sizes, int n_in,
                              void* d_out, int out_size, void* d_ws, size_t ws_size,
                              hipStream_t stream) {
  (void)in_sizes; (void)n_in; (void)out_size; (void)ws_size;
  const float* in = (const float*)d_in[0];   // [32,256,32,32] fp32 NCHW
  const float* w  = (const float*)d_in[1];   // [1024,256] fp32
  float* out = (float*)d_out;                // [32,256,32,32] fp32 + loss scalar

  char* ws = (char*)d_ws;                                    // ~33.3 MB used
  unsigned short* xhi = (unsigned short*)ws;                                  // 16 MB
  unsigned short* xlo = (unsigned short*)(ws + (size_t)16 * 1024 * 1024);     // 16 MB
  unsigned short* whi = (unsigned short*)(ws + (size_t)32 * 1024 * 1024);     // 512 KB
  unsigned short* wlo = (unsigned short*)(ws + (size_t)32 * 1024 * 1024 + 512 * 1024);
  float* hwn = (float*)(ws + (size_t)33 * 1024 * 1024);                       // 4 KB
  unsigned long long* best = (unsigned long long*)(ws + (size_t)33 * 1024 * 1024 + 64 * 1024); // 256 KB
  float* part = (float*)(ws + (size_t)33 * 1024 * 1024 + 64 * 1024 + 256 * 1024);             // 4 KB
  float* loss_slot = out + NDTOT;

  vq_prep_w<<<NCODES, 256, 0, stream>>>(w, whi, wlo, hwn, best);
  vq_prep_x<<<1024, 256, 0, stream>>>(in, xhi, xlo);
  vq_gemm<<<dim3(NCODES / 128, NROWS / 128), 256, 0, stream>>>(xhi, xlo, whi, wlo, hwn, best);
  vq_out<<<1024, 256, 0, stream>>>(in, w, best, out, part);
  vq_loss<<<1, 256, 0, stream>>>(part, loss_slot);
}
